// SelfAttention_10264971837797
// MI455X (gfx1250) — matrix operand order repaired and checked by reading, not compiled, not run on hardware
//
#include <hip/hip_runtime.h>
#include <stdint.h>

typedef __bf16 bf16_t;
typedef __attribute__((ext_vector_type(16))) __bf16 v16bf;
typedef __attribute__((ext_vector_type(8)))  __bf16 v8bf;
typedef __attribute__((ext_vector_type(8)))  float   v8f;

#define B_ 8
#define L_ 2048
#define D_ 256
#define QT 64            // query rows per block (4 waves x 16)
#define KT 32            // keys per tile
#define NWAVES 4
#define KVA_STRIDE 264   // padded bf16 row stride for [32 x 256] row-major KV tile
#define KVB_STRIDE 40    // padded bf16 row stride for [256 x 32] transposed KV tile
#define PB_STRIDE  40    // padded bf16 row stride for per-wave P buffer [16 x 32]

__device__ __forceinline__ v16bf ld16(const bf16_t* p) {
  v8bf lo = *(const v8bf*)p;
  v8bf hi = *(const v8bf*)(p + 8);
  v16bf r;
#pragma unroll
  for (int i = 0; i < 8; ++i) { r[i] = lo[i]; r[i + 8] = hi[i]; }
  return r;
}

__global__ __launch_bounds__(128)
void fa_fwd_gfx1250(const float* __restrict__ inp, const float* __restrict__ mem,
                    const int* __restrict__ mask, const float* __restrict__ wlin,
                    const float* __restrict__ dscale, float* __restrict__ out)
{
  // f32 staging for async global->LDS copy (double buffered, raw tile bytes)
  __shared__ __align__(16) float  stage[2][KT][D_];
  __shared__ __align__(16) bf16_t kvA[KT][KVA_STRIDE];          // [key][d] row-major
  __shared__ __align__(16) bf16_t kvB[D_][KVB_STRIDE];          // [d][key] transposed
  __shared__ __align__(16) bf16_t qfrag[NWAVES][8][32][16];     // Q A-frags, per-lane contiguous
  __shared__ __align__(16) bf16_t pbuf[NWAVES][16][PB_STRIDE];
  __shared__ float idotb[NWAVES][16];
  __shared__ int   maskb[NWAVES][16];

  const int tid  = threadIdx.x;
  const int lane = tid & 31;
  const int w    = tid >> 5;
  const int ll   = lane & 15;   // column within half-wave
  const int hh   = lane >> 4;   // half-wave select

  const int b  = blockIdx.x >> 5;          // L_/QT == 32 q-tiles per batch
  const int q0 = (blockIdx.x & 31) * QT;
  const int qw = q0 + w * 16;              // this wave's first query row

  const float* mbase = mem + (size_t)b * L_ * D_;
  const uint32_t stage_base = (uint32_t)(uintptr_t)(&stage[0][0][0]);

  // ---- async copy of one 32x256 f32 tile into stage[buf] (ASYNCcnt path) ----
  auto issue_async = [&](int kb2, int buf) {
    const float* gsrc = mbase + (size_t)kb2 * D_;
    uint32_t lbase = stage_base + (uint32_t)buf * (KT * D_ * 4);
#pragma unroll
    for (int i = 0; i < 16; ++i) {
      int j = tid + i * 128;               // float4 index within tile
      const float* ga = gsrc + j * 4;
      uint32_t la = lbase + (uint32_t)(j * 16);
      asm volatile("global_load_async_to_lds_b128 %0, %1, off"
                   :: "v"(la), "v"(ga) : "memory");
    }
  };

  // kick off tile 0 immediately; overlaps with all the setup below
  issue_async(0, 0);

  // ---------- pass-through: out[..., 0:256] = input (coalesced float4) ----------
  {
    const float4* in4 = (const float4*)inp;
#pragma unroll
    for (int i = 0; i < 32; ++i) {
      int j  = tid + i * 128;              // 64 rows * 64 float4s
      int qr = j >> 6;
      int d4 = j & 63;
      float4 v = in4[(size_t)(b * L_ + q0 + qr) * (D_ / 4) + d4];
      *(float4*)(out + (size_t)(b * L_ + q0 + qr) * (2 * D_) + d4 * 4) = v;
    }
  }

  // ---------- input_dot (Linear D->1) + mask for this wave's 16 rows ----------
  {
    const float* prow = inp + (size_t)(b * L_ + qw + ll) * D_ + hh * 128;
    const float* pw   = wlin + hh * 128;
    float s = 0.f;
#pragma unroll
    for (int d = 0; d < 128; d += 4) {
      float4 a  = *(const float4*)(prow + d);
      float4 wv = *(const float4*)(pw + d);
      s += a.x * wv.x + a.y * wv.y + a.z * wv.z + a.w * wv.w;
    }
    s += __shfl_xor(s, 16, 32);
    if (lane < 16) {
      idotb[w][ll] = s;
      maskb[w][ll] = mask[b * L_ + qw + ll];
    }
  }

  // ---------- Q A-fragments -> LDS (contiguous k order k = 16*hh + s, per-lane contiguous) ----------
  {
    const float* qrow = inp + (size_t)(b * L_ + qw + ll) * D_;
#pragma unroll
    for (int c = 0; c < 8; ++c) {
      int d0 = c * 32 + hh * 16;           // K = 0..7 (or 16..23 for upper half)
      int d1 = d0 + 8;                     // K = 8..15 (or 24..31)
      float qv[16], sv[16];
      *(float4*)&qv[0]  = *(const float4*)(qrow + d0);
      *(float4*)&qv[4]  = *(const float4*)(qrow + d0 + 4);
      *(float4*)&qv[8]  = *(const float4*)(qrow + d1);
      *(float4*)&qv[12] = *(const float4*)(qrow + d1 + 4);
      *(float4*)&sv[0]  = *(const float4*)(dscale + d0);
      *(float4*)&sv[4]  = *(const float4*)(dscale + d0 + 4);
      *(float4*)&sv[8]  = *(const float4*)(dscale + d1);
      *(float4*)&sv[12] = *(const float4*)(dscale + d1 + 4);
      bf16_t* qp = &qfrag[w][c][lane][0];
#pragma unroll
      for (int i = 0; i < 16; ++i) qp[i] = (bf16_t)(qv[i] * sv[i]);
    }
  }
  __syncthreads();

  // per-(lane,vgpr) row constants: x = s * rowscale + rowadd  (masked row => constant 0 logits)
  float rowadd[8], rowscale[8];
#pragma unroll
  for (int v = 0; v < 8; ++v) {
    int r = hh * 8 + v;
    int mz = (maskb[w][r] == 0);
    rowscale[v] = mz ? 0.f : 1.f;
    rowadd[v]   = mz ? 0.f : idotb[w][r];
  }

  // ---------- online-softmax state ----------
  v8f acc[16];
#pragma unroll
  for (int f = 0; f < 16; ++f) {
    v8f z = {0.f, 0.f, 0.f, 0.f, 0.f, 0.f, 0.f, 0.f};
    acc[f] = z;
  }
  float m_i[8], l_i[8];
#pragma unroll
  for (int v = 0; v < 8; ++v) { m_i[v] = -3.0e38f; l_i[v] = 0.f; }

  int cur = 0;
  for (int kb = 0; kb < L_; kb += KT) {
    // wait for this wave's async copies, then make all waves' copies visible;
    // this barrier also guarantees everyone finished compute on kvA/kvB.
    asm volatile("s_wait_asynccnt 0" ::: "memory");
    __syncthreads();

    // ---- convert staged f32 tile -> bf16 row-major (kvA) + transposed (kvB) ----
#pragma unroll
    for (int i = 0; i < 16; ++i) {
      int j  = tid + i * 128;
      int r  = j >> 6;                     // key row 0..31
      int dc = (j & 63) * 4;               // d column
      float4 vv = *(const float4*)&stage[cur][r][dc];
      bf16_t b0 = (bf16_t)vv.x, b1 = (bf16_t)vv.y;
      bf16_t b2 = (bf16_t)vv.z, b3 = (bf16_t)vv.w;
      kvA[r][dc + 0] = b0; kvA[r][dc + 1] = b1;
      kvA[r][dc + 2] = b2; kvA[r][dc + 3] = b3;
      kvB[dc + 0][r] = b0; kvB[dc + 1][r] = b1;
      kvB[dc + 2][r] = b2; kvB[dc + 3][r] = b3;
    }
    __syncthreads();

    // start fetching the next tile into the other buffer (overlaps with WMMAs)
    if (kb + KT < L_) issue_async(kb + KT, cur ^ 1);

    // ---- S = Q K^T : two 16x16 f32 fragments over 8 d-chunks of 32 ----
    v8f s0 = {0.f,0.f,0.f,0.f,0.f,0.f,0.f,0.f};
    v8f s1 = {0.f,0.f,0.f,0.f,0.f,0.f,0.f,0.f};
#pragma unroll
    for (int c = 0; c < 8; ++c) {
      v16bf aqc = ld16(&qfrag[w][c][lane][0]);
      v16bf bk0 = ld16(&kvA[ll     ][c * 32 + hh * 16]);   // cols = keys kb+0..15
      v16bf bk1 = ld16(&kvA[16 + ll][c * 32 + hh * 16]);   // cols = keys kb+16..31
      s0 = __builtin_amdgcn_wmma_f32_16x16x32_bf16(false, aqc, false, bk0,
                                                   (short)0, s0, false, false);
      s1 = __builtin_amdgcn_wmma_f32_16x16x32_bf16(false, aqc, false, bk1,
                                                   (short)0, s1, false, false);
    }

    // ---- online softmax over these 32 keys; P written straight to LDS ----
    float scl[8];
#pragma unroll
    for (int v = 0; v < 8; ++v) {
      float x0 = fmaf(s0[v], rowscale[v], rowadd[v]);
      float x1 = fmaf(s1[v], rowscale[v], rowadd[v]);
      float mx = fmaxf(x0, x1);
      mx = fmaxf(mx, __shfl_xor(mx, 1, 32));
      mx = fmaxf(mx, __shfl_xor(mx, 2, 32));
      mx = fmaxf(mx, __shfl_xor(mx, 4, 32));
      mx = fmaxf(mx, __shfl_xor(mx, 8, 32));
      float mnew = fmaxf(m_i[v], mx);
      float sc   = __expf(m_i[v] - mnew);
      float p0   = __expf(x0 - mnew);
      float p1   = __expf(x1 - mnew);
      int r = hh * 8 + v;
      pbuf[w][r][ll]      = (bf16_t)p0;
      pbuf[w][r][ll + 16] = (bf16_t)p1;
      float rs = p0 + p1;
      rs += __shfl_xor(rs, 1, 32);
      rs += __shfl_xor(rs, 2, 32);
      rs += __shfl_xor(rs, 4, 32);
      rs += __shfl_xor(rs, 8, 32);
      l_i[v] = l_i[v] * sc + rs;
      m_i[v] = mnew;
      scl[v] = sc;
    }

    // ---- rescale O accumulators ----
#pragma unroll
    for (int f = 0; f < 16; ++f)
#pragma unroll
      for (int v = 0; v < 8; ++v)
        acc[f][v] *= scl[v];

    // ---- P: C layout -> A layout via per-wave LDS (DS in-order within wave) ----
    asm volatile("s_wait_dscnt 0" ::: "memory");
    v16bf pa;
    {
      const bf16_t* pp = &pbuf[w][ll][hh * 16];
      v8bf lo = *(const v8bf*)pp;          // K = 0..7   (16..23 upper half)
      v8bf hi = *(const v8bf*)(pp + 8);    // K = 8..15  (24..31 upper half)
#pragma unroll
      for (int i = 0; i < 8; ++i) { pa[i] = lo[i]; pa[i + 8] = hi[i]; }
    }

    // ---- O += P V : 16 d-fragments, contraction over 32 keys ----
#pragma unroll
    for (int f = 0; f < 16; ++f) {
      v16bf bv = ld16(&kvB[f * 16 + ll][hh * 16]);
      acc[f] = __builtin_amdgcn_wmma_f32_16x16x32_bf16(false, pa, false, bv,
                                                       (short)0, acc[f], false, false);
    }
    cur ^= 1;
  }

  // ---------- epilogue: normalize and store out[..., 256:512] ----------
  float inv[8];
#pragma unroll
  for (int v = 0; v < 8; ++v) inv[v] = 1.0f / l_i[v];
#pragma unroll
  for (int f = 0; f < 16; ++f) {
#pragma unroll
    for (int v = 0; v < 8; ++v) {
      int r = hh * 8 + v;
      out[(size_t)(b * L_ + qw + r) * (2 * D_) + D_ + f * 16 + ll] = acc[f][v] * inv[v];
    }
  }
}

extern "C" void kernel_launch(void* const* d_in, const int* in_sizes, int n_in,
                              void* d_out, int out_size, void* d_ws, size_t ws_size,
                              hipStream_t stream) {
  const float* inp  = (const float*)d_in[0];
  const float* memp = (const float*)d_in[1];
  const int*   mk   = (const int*)d_in[2];
  const float* wlin = (const float*)d_in[3];
  const float* dsc  = (const float*)d_in[4];
  float* out = (float*)d_out;
  dim3 grid(B_ * (L_ / QT));   // 256 blocks
  dim3 block(128);             // 4 wave32s
  hipLaunchKernelGGL(fa_fwd_gfx1250, grid, block, 0, stream,
                     inp, memp, mk, wlin, dsc, out);
}
